// InteractionBlock_4922032521429
// MI455X (gfx1250) — matrix operand order repaired and checked by reading, not compiled, not run on hardware
//
#include <hip/hip_runtime.h>
#include <hip/hip_bf16.h>
#include <math.h>

// ---------------------------------------------------------------------------
// Problem constants (from the reference)
// ---------------------------------------------------------------------------
#define NN   50000      // nodes
#define EE   800000     // edges
#define HH   128        // hidden
#define MIDD 64
#define FF1  36
#define PEE  16

typedef __attribute__((ext_vector_type(16))) _Float16 v16h;
typedef __attribute__((ext_vector_type(8)))  float    v8f;

// ---------------------------------------------------------------------------
// WMMA helpers (V_WMMA_F32_16X16X32_F16, wave32, one 16x16 tile per wave)
// ---------------------------------------------------------------------------
__device__ __forceinline__ v8f wmma_f16(v16h a, v16h b, v8f c) {
  // 8 args: (neg_a, A, neg_b, B, c_mod, C, reuse_a, reuse_b)
  return __builtin_amdgcn_wmma_f32_16x16x32_f16(
      false, a, false, b, (short)0, c, false, false);
}

// A-matrix fragment (16x32, M on lanes, K contiguous per lane):
//   lanes 0-15 : M=lane,    K = k0+0..15
//   lanes16-31 : M=lane-16, K = k0+16..31
// Source: row-major fp32 matrix p[row][k], row stride ld. Zero-pad k>=kmax.
__device__ __forceinline__ v16h load_a_f32(const float* __restrict__ p, int ld,
                                           int row_base, int k0, int kmax) {
  const int l  = threadIdx.x & 31;
  const int m  = l & 15;
  const int kb = (l < 16) ? 0 : 16;
  const float* row = p + (size_t)(row_base + m) * ld;
  v16h a;
#pragma unroll
  for (int i = 0; i < 8; ++i) {
    int k1 = k0 + kb + i;
    int k2 = k0 + 8 + kb + i;
    a[i]     = (k1 < kmax) ? (_Float16)row[k1] : (_Float16)0.f;
    a[i + 8] = (k2 < kmax) ? (_Float16)row[k2] : (_Float16)0.f;
  }
  return a;
}

// B-matrix fragment (32x16, N on lanes, K contiguous per lane):
//   lanes 0-15 : N=lane,    K = k0+0..15
//   lanes16-31 : N=lane-16, K = k0+16..31
// B[k][n] = p[(col_base+n)*ld + coff + k]  (column n of B = row of p)
__device__ __forceinline__ v16h load_b_f32(const float* __restrict__ p, int ld,
                                           int col_base, int k0, int kmax,
                                           int coff) {
  const int l  = threadIdx.x & 31;
  const int n  = l & 15;
  const int kr = (l < 16) ? 0 : 16;
  const float* row = p + (size_t)(col_base + n) * ld + coff;
  v16h b;
#pragma unroll
  for (int i = 0; i < 16; ++i) {
    int k = k0 + kr + i;
    b[i] = (k < kmax) ? (_Float16)row[k] : (_Float16)0.f;
  }
  return b;
}

// B fragment from an LDS-staged f16 tile T[kdim][16] (kdim x 16, row-major)
__device__ __forceinline__ v16h load_b_lds(const _Float16* T, int k0) {
  const int l  = threadIdx.x & 31;
  const int e  = l & 15;
  const int kr = (l < 16) ? 0 : 16;
  v16h b;
#pragma unroll
  for (int i = 0; i < 16; ++i) b[i] = T[(k0 + kr + i) * 16 + e];
  return b;
}

__device__ __forceinline__ float swishf(float v) {
  return v / (1.f + __expf(-v));
}

// ---------------------------------------------------------------------------
// Generic node-level GEMM:  OUT[N,128] = epi( X1@W1^T (+ X2@W2^T) + b ) (+res)
// All X are [nrows,128] fp32 row-major; W row-major [128, ldw], col offset wc.
// One wave per 16x16 output tile; K=128 -> 4 WMMA k-steps per input.
// ---------------------------------------------------------------------------
#define FLAG_SWISH 1

__global__ __launch_bounds__(256)
void gemm_node_kernel(const float* __restrict__ X1, const float* __restrict__ W1,
                      int ldw1, int wc1,
                      const float* __restrict__ X2, const float* __restrict__ W2,
                      int ldw2, int wc2,
                      const float* __restrict__ bias,
                      const float* __restrict__ res,
                      float* __restrict__ OUT, int nrows, int flags) {
  const int wid = blockIdx.x * (blockDim.x >> 5) + (threadIdx.x >> 5);
  const int ntiles = (nrows >> 4) * (HH >> 4);
  if (wid >= ntiles) return;
  const int ct = wid & 7;           // column tile (8 tiles of 16 over H=128)
  const int rt = wid >> 3;          // row tile
  const int rbase = rt << 4, cbase = ct << 4;

  v8f acc = {};
#pragma unroll
  for (int ks = 0; ks < 128; ks += 32) {
    v16h a = load_a_f32(X1, HH, rbase, ks, 128);
    v16h b = load_b_f32(W1, ldw1, cbase, ks, 128, wc1);
    acc = wmma_f16(a, b, acc);
  }
  if (X2) {
#pragma unroll
    for (int ks = 0; ks < 128; ks += 32) {
      v16h a = load_a_f32(X2, HH, rbase, ks, 128);
      v16h b = load_b_f32(W2, ldw2, cbase, ks, 128, wc2);
      acc = wmma_f16(a, b, acc);
    }
  }

  // D layout: vgpr r -> M = r + (lane>=16 ? 8 : 0), N = lane&15
  const int l = threadIdx.x & 31;
  const int n = l & 15;
  const int moff = (l < 16) ? 0 : 8;
  const float bn = bias ? bias[cbase + n] : 0.f;
#pragma unroll
  for (int r = 0; r < 8; ++r) {
    const int row = rbase + r + moff;
    float v = acc[r] + bn;
    if (flags & FLAG_SWISH) v = swishf(v);
    if (res) v += res[(size_t)row * HH + cbase + n];
    OUT[(size_t)row * HH + cbase + n] = v;
  }
}

// ---------------------------------------------------------------------------
// Fused edge pipeline. One wave handles 16 edges:
//   T  = f1_w1[64,36] @ feat^T[36,16]      (WMMA, staged to LDS as f16)
//   F1 = f1_w2[128,64] @ T[64,16]          (WMMA, stays in registers)
//   agg1[dst] += F1 * x_lin_1[src]          (gather + atomic scatter epilogue)
// then the same with pos_emb / f2_* into agg2. f1/f2 never touch HBM.
// ---------------------------------------------------------------------------
__global__ __launch_bounds__(256)
void edge_fused_kernel(const float* __restrict__ feature1,
                       const float* __restrict__ pos_emb,
                       const int* __restrict__ src, const int* __restrict__ dst,
                       const float* __restrict__ f1_w1, const float* __restrict__ f1_w2,
                       const float* __restrict__ f2_w1, const float* __restrict__ f2_w2,
                       const float* __restrict__ xlin1,
                       float* __restrict__ agg1, float* __restrict__ agg2) {
  __shared__ _Float16 Tlds[8][MIDD * 16];   // per-wave staging tile, 2KB each
  const int wv  = threadIdx.x >> 5;
  const int wid = blockIdx.x * 8 + wv;
  if (wid * 16 >= EE) return;
  const int eb = wid * 16;

  const int l = threadIdx.x & 31;
  const int e = l & 15;
  const int moff = (l < 16) ? 0 : 8;
  const int s = src[eb + e];
  const int d = dst[eb + e];
  _Float16* T = &Tlds[wv][0];

  // prefetch next tile's features while we compute (speculative, RT)
  if (eb + 16 < EE) __builtin_prefetch(feature1 + (size_t)(eb + 16) * FF1, 0, 1);

  // ===================== feature1 branch (K=36) ============================
#pragma unroll
  for (int mt = 0; mt < 4; ++mt) {           // MID=64 -> 4 row tiles
    v8f acc = {};
#pragma unroll
    for (int ks = 0; ks < 64; ks += 32) {    // K=36 zero-padded to 64
      v16h a = load_a_f32(f1_w1, FF1, mt * 16, ks, FF1);
      v16h b = load_b_f32(feature1, FF1, eb, ks, FF1, 0);
      acc = wmma_f16(a, b, acc);
    }
#pragma unroll
    for (int r = 0; r < 8; ++r)
      T[(mt * 16 + r + moff) * 16 + e] = (_Float16)acc[r];
  }

#pragma unroll
  for (int ht = 0; ht < 8; ++ht) {           // H=128 -> 8 row tiles
    v8f acc = {};
#pragma unroll
    for (int ks = 0; ks < 64; ks += 32) {
      v16h a = load_a_f32(f1_w2, MIDD, ht * 16, ks, MIDD);
      v16h b = load_b_lds(T, ks);
      acc = wmma_f16(a, b, acc);
    }
    const float* xr = xlin1 + (size_t)s * HH + ht * 16 + moff;
    float* ar = agg1 + (size_t)d * HH + ht * 16 + moff;
#pragma unroll
    for (int r = 0; r < 8; ++r) atomicAdd(&ar[r], acc[r] * xr[r]);
  }

  // ===================== pos_emb branch (K=16) =============================
#pragma unroll
  for (int mt = 0; mt < 4; ++mt) {
    v8f acc = {};
    v16h a = load_a_f32(f2_w1, PEE, mt * 16, 0, PEE);   // K=16 padded to 32
    v16h b = load_b_f32(pos_emb, PEE, eb, 0, PEE, 0);
    acc = wmma_f16(a, b, acc);
#pragma unroll
    for (int r = 0; r < 8; ++r)
      T[(mt * 16 + r + moff) * 16 + e] = (_Float16)acc[r];
  }

#pragma unroll
  for (int ht = 0; ht < 8; ++ht) {
    v8f acc = {};
#pragma unroll
    for (int ks = 0; ks < 64; ks += 32) {
      v16h a = load_a_f32(f2_w2, MIDD, ht * 16, ks, MIDD);
      v16h b = load_b_lds(T, ks);
      acc = wmma_f16(a, b, acc);
    }
    const float* xr = xlin1 + (size_t)s * HH + ht * 16 + moff;
    float* ar = agg2 + (size_t)d * HH + ht * 16 + moff;
#pragma unroll
    for (int r = 0; r < 8; ++r) atomicAdd(&ar[r], acc[r] * xr[r]);
  }
}

// ---------------------------------------------------------------------------
// Host-side orchestration
// ---------------------------------------------------------------------------
extern "C" void kernel_launch(void* const* d_in, const int* in_sizes, int n_in,
                              void* d_out, int out_size, void* d_ws, size_t ws_size,
                              hipStream_t stream) {
  (void)in_sizes; (void)n_in; (void)out_size; (void)ws_size;

  const float* x        = (const float*)d_in[0];
  const float* feature1 = (const float*)d_in[1];
  const float* pos_emb  = (const float*)d_in[2];
  const int*   eidx     = (const int*)  d_in[3];   // [2,E]: row0=src, row1=dst
  const float* lin_1_w  = (const float*)d_in[5];
  const float* lin_1_b  = (const float*)d_in[6];
  const float* lin_2_w  = (const float*)d_in[7];
  const float* lin_2_b  = (const float*)d_in[8];
  const float* f1_w1    = (const float*)d_in[9];
  const float* f1_w2    = (const float*)d_in[10];
  const float* f2_w1    = (const float*)d_in[11];
  const float* f2_w2    = (const float*)d_in[12];
  const float* c1_ll_w  = (const float*)d_in[13];
  const float* c1_ll_b  = (const float*)d_in[14];
  const float* c1_lr_w  = (const float*)d_in[15];
  const float* c2_ll_w  = (const float*)d_in[16];
  const float* c2_ll_b  = (const float*)d_in[17];
  const float* c2_lr_w  = (const float*)d_in[18];
  const float* lin1_w   = (const float*)d_in[19];
  const float* lin1_b   = (const float*)d_in[20];
  const float* lin2_w   = (const float*)d_in[21];
  const float* lin2_b   = (const float*)d_in[22];
  const float* cat0_w   = (const float*)d_in[23];
  const float* cat0_b   = (const float*)d_in[24];
  const float* cat1_w   = (const float*)d_in[25];
  const float* cat1_b   = (const float*)d_in[26];
  const float* l0_w     = (const float*)d_in[27];
  const float* l0_b     = (const float*)d_in[28];
  const float* final_w  = (const float*)d_in[29];
  const float* final_b  = (const float*)d_in[30];

  const size_t NH = (size_t)NN * HH;
  float* B0 = (float*)d_ws;        // x_lin_1
  float* B1 = B0 + NH;             // x_lin_2
  float* B2 = B1 + NH;             // agg1 -> later h1
  float* B3 = B2 + NH;             // agg2 -> later h2
  float* B4 = B3 + NH;             // t1   -> later cat0 out
  float* B5 = B4 + NH;             // t2   -> later cat1 out

  const int src_off = 0, dst_off = EE;
  const dim3 blk(256);
  const int gemm_blocks = (NN / 16) * (HH / 16) / 8;   // 3125 blocks, 8 waves each
  const int edge_blocks = (EE / 16) / 8;               // 6250 blocks

  // 1) x_lin_1 = swish(x @ lin_1_w^T + b)
  hipLaunchKernelGGL(gemm_node_kernel, dim3(gemm_blocks), blk, 0, stream,
      x, lin_1_w, HH, 0, nullptr, nullptr, 0, 0, lin_1_b, nullptr, B0, NN, FLAG_SWISH);
  // 2) x_lin_2
  hipLaunchKernelGGL(gemm_node_kernel, dim3(gemm_blocks), blk, 0, stream,
      x, lin_2_w, HH, 0, nullptr, nullptr, 0, 0, lin_2_b, nullptr, B1, NN, FLAG_SWISH);

  // 3) zero scatter accumulators (agg1, agg2 are contiguous)
  hipMemsetAsync(B2, 0, 2 * NH * sizeof(float), stream);

  // 4) fused edge MLPs + gather + scatter-add
  hipLaunchKernelGGL(edge_fused_kernel, dim3(edge_blocks), blk, 0, stream,
      feature1, pos_emb, eidx + src_off, eidx + dst_off,
      f1_w1, f1_w2, f2_w1, f2_w2, B0, B2, B3);

  // 5) t1 = agg1 @ c1_ll^T + b + x_lin_1 @ c1_lr^T
  hipLaunchKernelGGL(gemm_node_kernel, dim3(gemm_blocks), blk, 0, stream,
      B2, c1_ll_w, HH, 0, B0, c1_lr_w, HH, 0, c1_ll_b, nullptr, B4, NN, 0);
  // 6) t2
  hipLaunchKernelGGL(gemm_node_kernel, dim3(gemm_blocks), blk, 0, stream,
      B3, c2_ll_w, HH, 0, B0, c2_lr_w, HH, 0, c2_ll_b, nullptr, B5, NN, 0);

  // 7) h1 = swish(t1 @ lin1^T + b)   (agg1 dead -> reuse B2)
  hipLaunchKernelGGL(gemm_node_kernel, dim3(gemm_blocks), blk, 0, stream,
      B4, lin1_w, HH, 0, nullptr, nullptr, 0, 0, lin1_b, nullptr, B2, NN, FLAG_SWISH);
  // 8) h2
  hipLaunchKernelGGL(gemm_node_kernel, dim3(gemm_blocks), blk, 0, stream,
      B5, lin2_w, HH, 0, nullptr, nullptr, 0, 0, lin2_b, nullptr, B3, NN, FLAG_SWISH);

  // 9) h = swish([h1|h2] @ cat0^T + b) — split cat0_w[128,256] into two K=128 passes
  hipLaunchKernelGGL(gemm_node_kernel, dim3(gemm_blocks), blk, 0, stream,
      B2, cat0_w, 2 * HH, 0, B3, cat0_w, 2 * HH, HH, cat0_b, nullptr, B4, NN, FLAG_SWISH);

  // 10) h = swish(h @ cat1^T + b) + x_lin_2
  hipLaunchKernelGGL(gemm_node_kernel, dim3(gemm_blocks), blk, 0, stream,
      B4, cat1_w, HH, 0, nullptr, nullptr, 0, 0, cat1_b, B1, B5, NN, FLAG_SWISH);

  // 11) h = swish(h @ l0^T + b)
  hipLaunchKernelGGL(gemm_node_kernel, dim3(gemm_blocks), blk, 0, stream,
      B5, l0_w, HH, 0, nullptr, nullptr, 0, 0, l0_b, nullptr, B0, NN, FLAG_SWISH);

  // 12) out = h @ final^T + b
  hipLaunchKernelGGL(gemm_node_kernel, dim3(gemm_blocks), blk, 0, stream,
      B0, final_w, HH, 0, nullptr, nullptr, 0, 0, final_b, nullptr,
      (float*)d_out, NN, 0);
}
